// LearnedSortOrder_34376918237593
// MI455X (gfx1250) — matrix operand order repaired and checked by reading, not compiled, not run on hardware
//
#include <hip/hip_runtime.h>
#include <hip/hip_bf16.h>

typedef __attribute__((ext_vector_type(16))) _Float16 v16h;
typedef __attribute__((ext_vector_type(8)))  float    v8f;

#define BIN 256
#define HID 32
#define CAPF 3.0f
#define NEGINF (-1e9f)
#define UPAD 260   // padded row stride for staged noise (bank-conflict avoidance)

__device__ __forceinline__ float gumbelf(float u) {
    // -log(-log(u + 1e-20) + 1e-20)
    return -__logf(-__logf(u + 1e-20f) + 1e-20f);
}

__device__ __forceinline__ unsigned lds_off(const void* p) {
    // generic LDS address: low 32 bits == offset from wave LDS base
    return (unsigned)(unsigned long long)p;
}

// ---------------------------------------------------------------------------
// Bulk kernel: items >= 768 all see a fully-saturated capacity mask (uniform
// -1e9), so they are independent. One block = 4 waves = 16 items x 256 bins.
// Wave w owns bins [w*64, w*64+64) as four 16x16 WMMA tiles; HIDDEN=32 means
// each logits tile is a single v_wmma_f32_16x16x32_f16. Noise rows are staged
// LDS-side via global_load_async_to_lds_b128 (ASYNCcnt), overlapped with the
// h-tile build + WMMA.
// ---------------------------------------------------------------------------
__global__ __launch_bounds__(128) void lso_bulk_kernel(
    const float* __restrict__ x,  const float* __restrict__ u1,
    const float* __restrict__ u2, const float* __restrict__ w1,
    const float* __restrict__ b1, const float* __restrict__ w2,
    const float* __restrict__ b2,
    float* __restrict__ out1, float* __restrict__ out2,
    int startTile, int nTiles)
{
    __shared__ _Float16 sW2t[BIN][HID + 2];   // w2 transposed, f16, padded
    __shared__ _Float16 sH[16][HID + 2];      // h tile (16 items x 32), f16
    __shared__ float sU1[16][UPAD];           // staged u1 rows (async copy)
    __shared__ float sU2[16][UPAD];           // staged u2 rows (async copy)
    __shared__ float sB2[BIN];
    __shared__ float sW1[HID], sB1[HID];
    __shared__ float sRedA[4][16];
    __shared__ float sRedB[4][16];
    __shared__ float sRedV[4][16];
    __shared__ int   sRedI[4][16];

    const int tid  = threadIdx.x;
    const int lane = tid & 31;
    const int wave = tid >> 5;    // 0..3
    const int g    = lane >> 4;   // lane half (K-split per ISA layout)
    const int n    = lane & 15;   // N index within tile / M index for A

    // Stage weights once per block (w2 is 32KB f32 -> 16KB f16, L2-resident).
    for (int i = tid; i < HID * BIN; i += blockDim.x) {
        int k = i >> 8, c = i & 255;
        sW2t[c][k] = (_Float16)w2[i];
    }
    for (int i = tid; i < BIN; i += blockDim.x) sB2[i] = b2[i];
    if (tid < HID) { sW1[tid] = w1[tid]; sB1[tid] = b1[tid]; }
    __syncthreads();

    // B operands are loop-invariant: B[K][N], lanes0-15 K=0..15, lanes16-31 K=16..31.
    v16h bmat[4];
    #pragma unroll
    for (int t = 0; t < 4; ++t) {
        const int n0 = wave * 64 + t * 16;
        #pragma unroll
        for (int h = 0; h < 16; ++h)
            bmat[t][h] = sW2t[n0 + n][g * 16 + h];
    }

    const int urow0 = tid >> 6;        // 0..1: two noise rows per copy round
    const int ucol  = (tid & 63) * 4;  // 0..252, 16B per lane

    for (int tile = startTile + blockIdx.x; tile < nTiles; tile += gridDim.x) {
        const int i0 = tile * 16;

        __syncthreads();   // prior-iteration LDS reads (sH, sU*) complete

        // Kick off async staging of this tile's noise (16 contiguous rows per
        // array). DMA runs while we build h and do the WMMAs.
        #pragma unroll
        for (int j = 0; j < 8; ++j) {
            const int m = j * 2 + urow0;
            const float* g1p = u1 + (size_t)(i0 + m) * BIN + ucol;
            const float* g2p = u2 + (size_t)(i0 + m) * BIN + ucol;
            const unsigned l1 = lds_off(&sU1[m][ucol]);
            const unsigned l2 = lds_off(&sU2[m][ucol]);
            asm volatile("global_load_async_to_lds_b128 %0, %1, off"
                         :: "v"(l1), "v"(g1p) : "memory");
            asm volatile("global_load_async_to_lds_b128 %0, %1, off"
                         :: "v"(l2), "v"(g2p) : "memory");
        }

        // h tile: h[m][k] = relu(x[i0+m]*w1[k] + b1[k])
        for (int i = tid; i < 16 * HID; i += blockDim.x) {
            int m = i >> 5, k = i & 31;
            sH[m][k] = (_Float16)fmaxf(0.f, x[i0 + m] * sW1[k] + sB1[k]);
        }
        __syncthreads();

        // A operand: 16-bit A 16x32 layout — lane L holds M=L%16;
        // element h -> K = g*16 + h (contiguous, matching B).
        v16h a;
        #pragma unroll
        for (int h = 0; h < 16; ++h) {
            const int k = g * 16 + h;
            a[h] = sH[n][k];
        }

        // masked logits = relu(h@w2 + b2) + NEGINF (capacity fully saturated)
        float msk[4][8];
        #pragma unroll
        for (int t = 0; t < 4; ++t) {
            v8f c = {};
            c = __builtin_amdgcn_wmma_f32_16x16x32_f16(
                    false, a, false, bmat[t], (short)0, c, false, false);
            const int n0 = wave * 64 + t * 16;
            #pragma unroll
            for (int r = 0; r < 8; ++r)
                msk[t][r] = fmaxf(0.f, c[r] + sB2[n0 + n]) + NEGINF;
        }

        // Drain async copies, then publish the staged noise to all waves.
        asm volatile("s_wait_asynccnt 0x0" ::: "memory");
        __syncthreads();

        // ---- p1 = softmax(msk + gumbel(u1)) over 256 bins per item ----
        // C/D layout: VGPR r holds item m = g*8 + r; bins stripe over lanes.
        float t1[4][8];
        #pragma unroll
        for (int t = 0; t < 4; ++t)
            #pragma unroll
            for (int r = 0; r < 8; ++r)
                t1[t][r] = msk[t][r] +
                           gumbelf(sU1[g * 8 + r][wave * 64 + t * 16 + n]);
        // wave-local max (reduce over 4 tiles + 16 lanes of this half-wave)
        #pragma unroll
        for (int r = 0; r < 8; ++r) {
            float v = fmaxf(fmaxf(t1[0][r], t1[1][r]), fmaxf(t1[2][r], t1[3][r]));
            #pragma unroll
            for (int s = 1; s <= 8; s <<= 1) v = fmaxf(v, __shfl_xor(v, s, 32));
            if (n == 0) sRedA[wave][g * 8 + r] = v;
        }
        __syncthreads();
        float gmax[8], ginv[8];
        #pragma unroll
        for (int r = 0; r < 8; ++r) {
            const int m = g * 8 + r;
            gmax[r] = fmaxf(fmaxf(sRedA[0][m], sRedA[1][m]),
                            fmaxf(sRedA[2][m], sRedA[3][m]));
        }
        #pragma unroll
        for (int r = 0; r < 8; ++r) {
            float s = 0.f;
            #pragma unroll
            for (int t = 0; t < 4; ++t) { t1[t][r] = __expf(t1[t][r] - gmax[r]); s += t1[t][r]; }
            #pragma unroll
            for (int m = 1; m <= 8; m <<= 1) s += __shfl_xor(s, m, 32);
            if (n == 0) sRedB[wave][g * 8 + r] = s;
        }
        __syncthreads();
        #pragma unroll
        for (int r = 0; r < 8; ++r) {
            const int m = g * 8 + r;
            ginv[r] = 1.f / (sRedB[0][m] + sRedB[1][m] + sRedB[2][m] + sRedB[3][m]);
        }
        #pragma unroll
        for (int t = 0; t < 4; ++t)
            #pragma unroll
            for (int r = 0; r < 8; ++r) {
                const int item = i0 + g * 8 + r;
                const int bin  = wave * 64 + t * 16 + n;
                out1[(size_t)item * BIN + bin] = t1[t][r] * ginv[r];
            }

        // ---- p2 = one_hot(argmax(msk + gumbel(u2))) (straight-through fwd) ----
        float bv[8]; int bi[8];
        #pragma unroll
        for (int r = 0; r < 8; ++r) { bv[r] = -3.4e38f; bi[r] = 0x7fffffff; }
        #pragma unroll
        for (int t = 0; t < 4; ++t)
            #pragma unroll
            for (int r = 0; r < 8; ++r) {
                const int bin = wave * 64 + t * 16 + n;
                const float v = msk[t][r] + gumbelf(sU2[g * 8 + r][bin]);
                if (v > bv[r]) { bv[r] = v; bi[r] = bin; }   // t ascending => lowest bin on ties
            }
        #pragma unroll
        for (int r = 0; r < 8; ++r) {
            #pragma unroll
            for (int s = 1; s <= 8; s <<= 1) {
                const float ov = __shfl_xor(bv[r], s, 32);
                const int   oi = __shfl_xor(bi[r], s, 32);
                if (ov > bv[r] || (ov == bv[r] && oi < bi[r])) { bv[r] = ov; bi[r] = oi; }
            }
            if (n == 0) { sRedV[wave][g * 8 + r] = bv[r]; sRedI[wave][g * 8 + r] = bi[r]; }
        }
        __syncthreads();
        int gbi[8];
        #pragma unroll
        for (int r = 0; r < 8; ++r) {
            const int m = g * 8 + r;
            float vv = sRedV[0][m]; int ii = sRedI[0][m];
            #pragma unroll
            for (int w = 1; w < 4; ++w) {
                const float ov = sRedV[w][m]; const int oi = sRedI[w][m];
                if (ov > vv || (ov == vv && oi < ii)) { vv = ov; ii = oi; }
            }
            gbi[r] = ii;
        }
        #pragma unroll
        for (int t = 0; t < 4; ++t)
            #pragma unroll
            for (int r = 0; r < 8; ++r) {
                const int item = i0 + g * 8 + r;
                const int bin  = wave * 64 + t * 16 + n;
                out2[(size_t)item * BIN + bin] = (bin == gbi[r]) ? 1.f : 0.f;
            }
    }
}

// ---------------------------------------------------------------------------
// Sequential kernel: items 0..767 (until all 256 bins hit capacity 3).
// Single wave32, counts in registers (8 bins/lane), shuffle reductions only.
// ---------------------------------------------------------------------------
__global__ __launch_bounds__(32) void lso_seq_kernel(
    const float* __restrict__ x,  const float* __restrict__ u1,
    const float* __restrict__ u2, const float* __restrict__ w1,
    const float* __restrict__ b1, const float* __restrict__ w2,
    const float* __restrict__ b2,
    float* __restrict__ out1, float* __restrict__ out2, int seqN)
{
    __shared__ float sW2[HID][BIN];
    __shared__ float sW1[HID], sB1[HID], sB2[BIN];
    const int lane = threadIdx.x;

    for (int i = lane; i < HID * BIN; i += 32) sW2[i >> 8][i & 255] = w2[i];
    for (int i = lane; i < BIN; i += 32) sB2[i] = b2[i];
    if (lane < HID) { sW1[lane] = w1[lane]; sB1[lane] = b1[lane]; }
    __syncthreads();

    float cnt[8] = {0.f, 0.f, 0.f, 0.f, 0.f, 0.f, 0.f, 0.f};

    for (int i = 0; i < seqN; ++i) {
        if (i + 1 < seqN) {
            __builtin_prefetch(&u1[(size_t)(i + 1) * BIN + lane * 8], 0, 0);
            __builtin_prefetch(&u2[(size_t)(i + 1) * BIN + lane * 8], 0, 0);
        }
        const float xi = x[i];
        const float hv = fmaxf(0.f, xi * sW1[lane] + sB1[lane]);  // h[lane]

        float lg[8] = {0.f, 0.f, 0.f, 0.f, 0.f, 0.f, 0.f, 0.f};
        #pragma unroll 8
        for (int k = 0; k < HID; ++k) {
            const float hk = __shfl(hv, k, 32);
            #pragma unroll
            for (int j = 0; j < 8; ++j)
                lg[j] = fmaf(hk, sW2[k][j * 32 + lane], lg[j]);
        }

        float t1[8], t2[8];
        #pragma unroll
        for (int j = 0; j < 8; ++j) {
            const int bin = j * 32 + lane;
            const float l = fmaxf(0.f, lg[j] + sB2[bin]);
            const float m = (cnt[j] >= CAPF) ? NEGINF : 0.f;
            t1[j] = l + m + gumbelf(u1[(size_t)i * BIN + bin]);
            t2[j] = l + m + gumbelf(u2[(size_t)i * BIN + bin]);
        }

        float mx = t1[0];
        #pragma unroll
        for (int j = 1; j < 8; ++j) mx = fmaxf(mx, t1[j]);
        #pragma unroll
        for (int s = 1; s <= 16; s <<= 1) mx = fmaxf(mx, __shfl_xor(mx, s, 32));
        float sum = 0.f;
        #pragma unroll
        for (int j = 0; j < 8; ++j) { t1[j] = __expf(t1[j] - mx); sum += t1[j]; }
        #pragma unroll
        for (int s = 1; s <= 16; s <<= 1) sum += __shfl_xor(sum, s, 32);
        const float inv = 1.f / sum;
        #pragma unroll
        for (int j = 0; j < 8; ++j)
            out1[(size_t)i * BIN + j * 32 + lane] = t1[j] * inv;

        float bv = t2[0]; int bi = lane;
        #pragma unroll
        for (int j = 1; j < 8; ++j) {
            const int bin = j * 32 + lane;
            if (t2[j] > bv) { bv = t2[j]; bi = bin; }
        }
        #pragma unroll
        for (int s = 1; s <= 16; s <<= 1) {
            const float ov = __shfl_xor(bv, s, 32);
            const int   oi = __shfl_xor(bi, s, 32);
            if (ov > bv || (ov == bv && oi < bi)) { bv = ov; bi = oi; }
        }
        #pragma unroll
        for (int j = 0; j < 8; ++j) {
            const int bin = j * 32 + lane;
            const float one = (bin == bi) ? 1.f : 0.f;
            out2[(size_t)i * BIN + bin] = one;
            cnt[j] += one;
        }
    }
}

extern "C" void kernel_launch(void* const* d_in, const int* in_sizes, int n_in,
                              void* d_out, int out_size, void* d_ws, size_t ws_size,
                              hipStream_t stream) {
    const float* x  = (const float*)d_in[0];
    const float* u1 = (const float*)d_in[1];
    const float* u2 = (const float*)d_in[2];
    const float* w1 = (const float*)d_in[3];
    const float* b1 = (const float*)d_in[4];
    const float* w2 = (const float*)d_in[5];
    const float* b2 = (const float*)d_in[6];
    const int N = in_sizes[0];           // x is [N,1]
    float* out1 = (float*)d_out;
    float* out2 = out1 + (size_t)N * BIN;

    int seqN = 3 * BIN;                  // capacity saturation point (768)
    if (seqN > N) seqN = N;
    const int nTiles    = N / 16;
    const int startTile = seqN / 16;

    if (startTile < nTiles) {
        int blocks = nTiles - startTile;
        if (blocks > 2048) blocks = 2048;
        lso_bulk_kernel<<<blocks, 128, 0, stream>>>(
            x, u1, u2, w1, b1, w2, b2, out1, out2, startTile, nTiles);
    }
    if (seqN > 0) {
        lso_seq_kernel<<<1, 32, 0, stream>>>(
            x, u1, u2, w1, b1, w2, b2, out1, out2, seqN);
    }
}
